// SparseLinear_61761629716951
// MI455X (gfx1250) — matrix operand order repaired and checked by reading, not compiled, not run on hardware
//
#include <hip/hip_runtime.h>

// Block-sparse linear (tiled_regular mask) as gather-GEMM per output block:
//   out-block c = sum_{k=0..7} Xblk[(c-16k) mod 128] @ W[((c-16k)mod128)*8 + k]
// fp32 fidelity via split-bf16 (hi/lo truncation split, exact residual):
//   x*w ~= xh*wh + xh*wl + xl*wh   (3 bf16 WMMAs per K=32, fp32 accumulate)
// Double-buffered LDS overlaps next block's HBM loads with current WMMAs.
// Wave tiling: 8 waves = 4(M) x 2(N), each wave owns a 32x32 C tile so both
// A and B fragments are reused by 2 tiles -> 32 ds_load_b128 : 24 wmma / block.

#define N_DIM   8192
#define BS      64
#define NB      128
#define OCC     8
#define BSTRIDE 16     // NB / OCC
#define MTILE   128    // batch rows per workgroup
#define XS      72     // LDS row stride (bf16 elems): 144B = 36 banks, conflict-free
#define WS      72

typedef __attribute__((ext_vector_type(16))) __bf16   v16bf;
typedef __attribute__((ext_vector_type(8)))  __bf16   v8bf;
typedef __attribute__((ext_vector_type(8)))  float    v8f;
typedef __attribute__((ext_vector_type(4)))  float    v4f;
typedef __attribute__((ext_vector_type(2)))  unsigned v2u;

union Frag16 { v16bf v16; v8bf v8[2]; };

// Truncation split of two f32 into packed hi-bf16 pair (.x) and lo-bf16 pair (.y).
// hi = top 16 bits (residual is exact); lo = top 16 bits of residual.
static __device__ __forceinline__ v2u split2(float a, float b) {
  const unsigned ua = __builtin_bit_cast(unsigned, a);
  const unsigned ub = __builtin_bit_cast(unsigned, b);
  const unsigned hi = (ua >> 16) | (ub & 0xffff0000u);
  const float ra = a - __builtin_bit_cast(float, ua & 0xffff0000u);
  const float rb = b - __builtin_bit_cast(float, ub & 0xffff0000u);
  const unsigned lo = (__builtin_bit_cast(unsigned, ra) >> 16) |
                      (__builtin_bit_cast(unsigned, rb) & 0xffff0000u);
  v2u r = { hi, lo };
  return r;
}

__global__ __launch_bounds__(256)
void sparse_linear_wmma(const float* __restrict__ x,
                        const float* __restrict__ w,
                        float* __restrict__ out) {
  __shared__ __attribute__((aligned(16))) unsigned short sXh[2][MTILE * XS];
  __shared__ __attribute__((aligned(16))) unsigned short sXl[2][MTILE * XS];
  __shared__ __attribute__((aligned(16))) unsigned short sWh[2][BS * WS];
  __shared__ __attribute__((aligned(16))) unsigned short sWl[2][BS * WS];

  const int tid    = threadIdx.x;
  const int cblk   = blockIdx.x;          // output block column, 0..127
  const int batch0 = blockIdx.y * MTILE;

  const int wave  = tid >> 5;             // wave32: 8 waves = 4(M) x 2(N)
  const int lane  = tid & 31;
  const int lrow  = lane & 15;
  const int lhalf = lane >> 4;
  const int mw    = wave & 3;             // M tile group: rows mw*32..mw*32+31
  const int nw    = wave >> 2;            // N half:      cols nw*32..nw*32+31

  v8f acc[4];                             // [mt*2 + jt]
#pragma unroll
  for (int j = 0; j < 4; ++j) acc[j] = (v8f)0.0f;

  v4f xr[8];   // staged X tile slice (this thread's 8 float4)
  v4f wr[4];   // staged W 4x4 sub-tile (4 rows of k, 4 cols of n)

  // ---- issue global loads for block-step kb into registers ----
  auto load_global = [&](int kb) {
    const int rblk = (cblk + NB - kb * BSTRIDE) & (NB - 1);
    const int nidx = rblk * OCC + kb;
    const float* xg = x + (size_t)batch0 * N_DIM + rblk * BS;
    const float* wg = w + (size_t)nidx * (BS * BS)
                        + (tid >> 4) * (4 * BS)   // krow0 * BS
                        + (tid & 15) * 4;         // no0
#pragma unroll
    for (int i = 0; i < 8; ++i) {
      const int idx = tid + i * 256;              // 0..2047 float4 slots
      xr[i] = *(const v4f*)(xg + (size_t)(idx >> 4) * N_DIM + (idx & 15) * 4);
    }
#pragma unroll
    for (int r = 0; r < 4; ++r)
      wr[r] = *(const v4f*)(wg + r * BS);
  };

  // ---- convert staged registers and store to LDS buffer `buf` ----
  auto stage_lds = [&](int buf) {
#pragma unroll
    for (int i = 0; i < 8; ++i) {
      const int idx = tid + i * 256;
      const int row = idx >> 4;
      const int kk  = (idx & 15) * 4;
      const v2u s01 = split2(xr[i][0], xr[i][1]);
      const v2u s23 = split2(xr[i][2], xr[i][3]);
      v2u ph = { s01.x, s23.x };
      v2u pl = { s01.y, s23.y };
      *(v2u*)&sXh[buf][row * XS + kk] = ph;
      *(v2u*)&sXl[buf][row * XS + kk] = pl;
    }
    // W: register-transpose the 4x4 sub-tile -> LDS [n][k], packed b64 stores
    const int krow0 = (tid >> 4) * 4;
    const int no0   = (tid & 15) * 4;
#pragma unroll
    for (int e = 0; e < 4; ++e) {
      const v2u s01 = split2(wr[0][e], wr[1][e]);
      const v2u s23 = split2(wr[2][e], wr[3][e]);
      v2u ph = { s01.x, s23.x };
      v2u pl = { s01.y, s23.y };
      *(v2u*)&sWh[buf][(no0 + e) * WS + krow0] = ph;
      *(v2u*)&sWl[buf][(no0 + e) * WS + krow0] = pl;
    }
  };

  // ---- WMMA over one staged block: K=64 as two K=32 steps, 3 products ----
  auto compute = [&](int buf) {
#pragma unroll
    for (int ks = 0; ks < 2; ++ks) {
      // A (16x32 bf16 layout): per-lane chunks at K = lhalf*8 and 16+lhalf*8
      const int ka0 = ks * 32 + lhalf * 8;
      const int ka1 = ks * 32 + 16 + lhalf * 8;
      Frag16 ah[2], al[2];
#pragma unroll
      for (int mt = 0; mt < 2; ++mt) {
        const int arow = mw * 32 + mt * 16 + lrow;
        ah[mt].v8[0] = *(const v8bf*)&sXh[buf][arow * XS + ka0];
        ah[mt].v8[1] = *(const v8bf*)&sXh[buf][arow * XS + ka1];
        al[mt].v8[0] = *(const v8bf*)&sXl[buf][arow * XS + ka0];
        al[mt].v8[1] = *(const v8bf*)&sXl[buf][arow * XS + ka1];
      }
      // B (32x16 bf16 layout): per-lane chunks at K = lhalf*8 and 16+lhalf*8
      const int kb0 = ks * 32 + lhalf * 8;
#pragma unroll
      for (int jt = 0; jt < 2; ++jt) {
        const int bn = nw * 32 + jt * 16 + lrow;
        Frag16 bh, bl;
        bh.v8[0] = *(const v8bf*)&sWh[buf][bn * WS + kb0];
        bh.v8[1] = *(const v8bf*)&sWh[buf][bn * WS + kb0 + 16];
        bl.v8[0] = *(const v8bf*)&sWl[buf][bn * WS + kb0];
        bl.v8[1] = *(const v8bf*)&sWl[buf][bn * WS + kb0 + 16];
#pragma unroll
        for (int mt = 0; mt < 2; ++mt) {
          const int a = mt * 2 + jt;
          acc[a] = __builtin_amdgcn_wmma_f32_16x16x32_bf16(
              false, ah[mt].v16, false, bh.v16, (short)0, acc[a], false, false);
          acc[a] = __builtin_amdgcn_wmma_f32_16x16x32_bf16(
              false, ah[mt].v16, false, bl.v16, (short)0, acc[a], false, false);
          acc[a] = __builtin_amdgcn_wmma_f32_16x16x32_bf16(
              false, al[mt].v16, false, bh.v16, (short)0, acc[a], false, false);
        }
      }
    }
  };

  // ---- software pipeline: global loads run under the WMMAs ----
  load_global(0);
  stage_lds(0);
  __syncthreads();

  for (int kb = 0; kb < OCC; ++kb) {
    if (kb + 1 < OCC) {
      load_global(kb + 1);                         // HBM latency hidden by WMMA
      if (kb + 2 < OCC) {                          // touch weights 2 ahead
        const int rn = (cblk + NB - (kb + 2) * BSTRIDE) & (NB - 1);
        const float* wn = w + (size_t)(rn * OCC + kb + 2) * (BS * BS);
        if (tid < 128) __builtin_prefetch(wn + tid * 32, 0, 3);
      }
    }
    compute(kb & 1);
    if (kb + 1 < OCC) stage_lds((kb + 1) & 1);     // fill the other buffer
    __syncthreads();
  }

  // ---- epilogue: C layout = VGPR v -> row (v + lhalf*8), col = lrow ----
  float* og = out + ((size_t)batch0 + mw * 32 + lhalf * 8) * N_DIM
                  + cblk * BS + nw * 32;
#pragma unroll
  for (int mt = 0; mt < 2; ++mt)
#pragma unroll
    for (int jt = 0; jt < 2; ++jt)
#pragma unroll
      for (int v = 0; v < 8; ++v)
        og[(size_t)(mt * 16 + v) * N_DIM + jt * 16 + lrow] = acc[mt * 2 + jt][v];
}

extern "C" void kernel_launch(void* const* d_in, const int* in_sizes, int n_in,
                              void* d_out, int out_size, void* d_ws, size_t ws_size,
                              hipStream_t stream) {
  const float* x = (const float*)d_in[0];     // (2048, 8192) f32
  const float* w = (const float*)d_in[1];     // (1024, 64, 64) f32
  float* out = (float*)d_out;                 // (2048, 8192) f32

  const int batch = in_sizes[0] / N_DIM;      // 2048
  dim3 grid(NB, batch / MTILE);               // 128 x 16 workgroups
  sparse_linear_wmma<<<grid, 256, 0, stream>>>(x, w, out);
}
